// SelfAttentionRPPEBlock_23330262352160
// MI455X (gfx1250) — compile-verified, matrix operand order checked by reading, not run on hardware
//
#include <hip/hip_runtime.h>
#include <hip/hip_bf16.h>

typedef __bf16 bf16;
typedef __attribute__((ext_vector_type(16))) __bf16 v16bf;
typedef __attribute__((ext_vector_type(8)))  __bf16 v8bf;
typedef __attribute__((ext_vector_type(4)))  __bf16 v4bf;
typedef __attribute__((ext_vector_type(8)))  float   v8f;

// ---------------------------------------------------------------- constants
constexpr int  Bn   = 8;
constexpr int  Tn   = 2048;
constexpr int  DIN  = 1024;
constexpr int  DEMB = 1024;
constexpr int  DOUT = 1024;
constexpr int  E3   = 3 * DEMB;          // 3072
constexpr long Mrows = (long)Bn * Tn;    // 16384

// workspace layout (bytes)
constexpr size_t OFF_XBF  = 0;                                    // x bf16       32 MB
constexpr size_t OFF_WQKV = OFF_XBF  + (size_t)Mrows * DIN * 2;   // Wqkv bf16     6 MB
constexpr size_t OFF_WOUT = OFF_WQKV + (size_t)E3 * DIN * 2;      // Wout bf16     2 MB
constexpr size_t OFF_QKV  = OFF_WOUT + (size_t)DOUT * DEMB * 2;   // qkv bf16     96 MB
constexpr size_t OFF_S    = OFF_QKV  + (size_t)Mrows * E3 * 2;    // S f32       128 MB
constexpr size_t OFF_P    = OFF_S    + (size_t)Bn * Tn * Tn * 4;  // P bf16       64 MB
constexpr size_t OFF_V    = OFF_P    + (size_t)Bn * Tn * Tn * 2;  // values bf16  32 MB
constexpr size_t OFF_VT   = OFF_V    + (size_t)Mrows * DEMB * 2;  // V^T bf16     32 MB

// ---------------------------------------------------------------- convert (vectorized)
__global__ __launch_bounds__(256) void f32_to_bf16(const float* __restrict__ in,
                                                   bf16* __restrict__ out, long n4) {
  long i = (long)blockIdx.x * blockDim.x + threadIdx.x;
  long stride = (long)gridDim.x * blockDim.x;
  for (; i < n4; i += stride) {
    float4 f = ((const float4*)in)[i];
    v4bf o = { (bf16)f.x, (bf16)f.y, (bf16)f.z, (bf16)f.w };
    ((v4bf*)out)[i] = o;
  }
}

// ---------------------------------------------------------------- V transpose (per batch)
// vT[b][d][s] = v[b][s][d]; v rows live inside qkv with stride E3
__global__ __launch_bounds__(256) void transpose_v(const bf16* __restrict__ v,
                                                   bf16* __restrict__ vt) {
  __shared__ bf16 tile[32][33];
  const int b  = blockIdx.z;
  const int s0 = blockIdx.x * 32;
  const int d0 = blockIdx.y * 32;
  const int tx = threadIdx.x & 31;
  const int ty = threadIdx.x >> 5;           // 0..7
  const bf16* vb = v  + (size_t)b * Tn * E3;
  bf16*       tb = vt + (size_t)b * DEMB * Tn;
#pragma unroll
  for (int i = 0; i < 4; ++i)                // read: contiguous in d
    tile[ty * 4 + i][tx] = vb[(size_t)(s0 + ty * 4 + i) * E3 + d0 + tx];
  __syncthreads();
#pragma unroll
  for (int i = 0; i < 4; ++i)                // write: contiguous in s
    tb[(size_t)(d0 + ty * 4 + i) * Tn + s0 + tx] = tile[tx][ty * 4 + i];
}

// ---------------------------------------------------------------- GEMM
// C[m,n] = sum_k A[m,k] * B[n,k]  (+ epilogue)
// A: row-major [M x K] lda;  B: row-major [N x K] ldb (K contiguous for both)
enum { EPI_QKV = 0, EPI_SCORES = 1, EPI_PLAIN = 2, EPI_OUT = 3 };

template <int EPI>
__global__ __launch_bounds__(128) void gemm_wmma(
    const bf16* __restrict__ A, long sA, int lda,
    const bf16* __restrict__ B, long sB, int ldb,
    void* __restrict__ C, long sC, int ldc,
    const float* __restrict__ bias, int ldbias, int K) {
  const int lane  = threadIdx.x & 31;
  const int wid   = threadIdx.x >> 5;
  const int rlane = lane & 15;
  const int koff  = (lane >= 16) ? 8 : 0;

  const long z = blockIdx.z;
  const bf16* Ab = A + z * sA;
  const bf16* Bb = B + z * sB;

  const int m0 = blockIdx.y * 128 + (wid >> 1) * 64;
  const int n0 = blockIdx.x * 128 + (wid & 1) * 64;

  v8f acc[4][4];
#pragma unroll
  for (int i = 0; i < 4; ++i)
#pragma unroll
    for (int j = 0; j < 4; ++j) acc[i][j] = (v8f)(0.0f);

  for (int k = 0; k < K; k += 32) {
    v16bf af[4], bfr[4];
#pragma unroll
    for (int mt = 0; mt < 4; ++mt) {
      const bf16* p = Ab + (size_t)(m0 + mt * 16 + rlane) * lda + k + koff;
      v8bf lo = *(const v8bf*)p;
      v8bf hi = *(const v8bf*)(p + 16);
      __builtin_prefetch(p + 64, 0, 1);      // unconditional: branch-free
#pragma unroll
      for (int i = 0; i < 8; ++i) { af[mt][i] = lo[i]; af[mt][i + 8] = hi[i]; }
    }
#pragma unroll
    for (int nt = 0; nt < 4; ++nt) {
      const bf16* p = Bb + (size_t)(n0 + nt * 16 + rlane) * ldb + k + koff;
      v8bf lo = *(const v8bf*)p;
      v8bf hi = *(const v8bf*)(p + 16);
#pragma unroll
      for (int i = 0; i < 8; ++i) { bfr[nt][i] = lo[i]; bfr[nt][i + 8] = hi[i]; }
    }
#pragma unroll
    for (int mt = 0; mt < 4; ++mt)
#pragma unroll
      for (int nt = 0; nt < 4; ++nt)
        acc[mt][nt] = __builtin_amdgcn_wmma_f32_16x16x32_bf16(
            false, af[mt], false, bfr[nt], (short)0, acc[mt][nt], false, false);
  }

  // epilogue: C/D layout -> VGPR r, lane l: M = r + 8*(l>>4), N = l&15
  float* Cf = (float*)C + z * sC;
  bf16*  Ch = (bf16*)C + z * sC;
#pragma unroll
  for (int mt = 0; mt < 4; ++mt) {
#pragma unroll
    for (int nt = 0; nt < 4; ++nt) {
      const int n     = n0 + nt * 16 + (lane & 15);
      const int mbase = m0 + mt * 16 + ((lane >> 4) & 1) * 8;
      v8f c = acc[mt][nt];
#pragma unroll
      for (int r = 0; r < 8; ++r) {
        const int m = mbase + r;
        float v = c[r];
        if (EPI == EPI_QKV) {
          v = (v + bias[n]) * ((n < DEMB) ? 0.03125f : 1.0f);  // 1/sqrt(1024) on Q
          Ch[(size_t)m * ldc + n] = (bf16)v;
        } else if (EPI == EPI_SCORES) {
          v += bias[(size_t)m * ldbias + n];                    // additive [T,T] bias
          Cf[(size_t)m * ldc + n] = v;
        } else if (EPI == EPI_PLAIN) {
          Ch[(size_t)m * ldc + n] = (bf16)v;
        } else {                                                // EPI_OUT
          Cf[(size_t)m * ldc + n] = v + bias[n];
        }
      }
    }
  }
}

// ---------------------------------------------------------------- softmax
__device__ inline float wave_max(float v) {
#pragma unroll
  for (int o = 16; o > 0; o >>= 1) v = fmaxf(v, __shfl_xor(v, o, 32));
  return v;
}
__device__ inline float wave_sum(float v) {
#pragma unroll
  for (int o = 16; o > 0; o >>= 1) v += __shfl_xor(v, o, 32);
  return v;
}

__global__ __launch_bounds__(256) void softmax_rows(const float* __restrict__ S,
                                                    bf16* __restrict__ P) {
  const long row = blockIdx.x;           // 0 .. B*T-1
  const int  tid = threadIdx.x;
  const int  wid = tid >> 5, lane = tid & 31;
  const float4* s4 = (const float4*)(S + row * (long)Tn);

  float4 a = s4[tid];
  float4 b = s4[tid + 256];
  float x[8] = {a.x, a.y, a.z, a.w, b.x, b.y, b.z, b.w};

  float m = x[0];
#pragma unroll
  for (int i = 1; i < 8; ++i) m = fmaxf(m, x[i]);
  m = wave_max(m);

  __shared__ float red[8];
  if (lane == 0) red[wid] = m;
  __syncthreads();
  float bm = red[0];
#pragma unroll
  for (int w = 1; w < 8; ++w) bm = fmaxf(bm, red[w]);

  float sum = 0.0f;
#pragma unroll
  for (int i = 0; i < 8; ++i) { x[i] = __expf(x[i] - bm); sum += x[i]; }
  sum = wave_sum(sum);
  __syncthreads();
  if (lane == 0) red[wid] = sum;
  __syncthreads();
  float bs = 0.0f;
#pragma unroll
  for (int w = 0; w < 8; ++w) bs += red[w];
  const float inv = 1.0f / bs;

  bf16* p = P + row * (long)Tn;
  v4bf o0 = { (bf16)(x[0] * inv), (bf16)(x[1] * inv), (bf16)(x[2] * inv), (bf16)(x[3] * inv) };
  v4bf o1 = { (bf16)(x[4] * inv), (bf16)(x[5] * inv), (bf16)(x[6] * inv), (bf16)(x[7] * inv) };
  *(v4bf*)(p + 4 * tid)        = o0;
  *(v4bf*)(p + 1024 + 4 * tid) = o1;
}

// ---------------------------------------------------------------- launch
extern "C" void kernel_launch(void* const* d_in, const int* in_sizes, int n_in,
                              void* d_out, int out_size, void* d_ws, size_t ws_size,
                              hipStream_t stream) {
  const float* x    = (const float*)d_in[0];
  const float* bias = (const float*)d_in[1];
  const float* Wqkv = (const float*)d_in[2];
  const float* bqkv = (const float*)d_in[3];
  const float* Wout = (const float*)d_in[4];
  const float* bout = (const float*)d_in[5];

  char* ws = (char*)d_ws;
  bf16*  x_bf    = (bf16*)(ws + OFF_XBF);
  bf16*  wqkv_bf = (bf16*)(ws + OFF_WQKV);
  bf16*  wout_bf = (bf16*)(ws + OFF_WOUT);
  bf16*  qkv_bf  = (bf16*)(ws + OFF_QKV);
  float* Sbuf    = (float*)(ws + OFF_S);
  bf16*  Pbuf    = (bf16*)(ws + OFF_P);
  bf16*  Vbuf    = (bf16*)(ws + OFF_V);
  bf16*  VTbuf   = (bf16*)(ws + OFF_VT);

  // 1) downconvert inputs to bf16 (float4 -> packed 4x bf16)
  f32_to_bf16<<<2048, 256, 0, stream>>>(x, x_bf, Mrows * DIN / 4);
  f32_to_bf16<<<512,  256, 0, stream>>>(Wqkv, wqkv_bf, (long)E3 * DIN / 4);
  f32_to_bf16<<<256,  256, 0, stream>>>(Wout, wout_bf, (long)DOUT * DEMB / 4);

  dim3 blk(128);

  // 2) fused QKV projection: [16384 x 1024] x [3072 x 1024]^T -> bf16 qkv, Q pre-scaled
  gemm_wmma<EPI_QKV><<<dim3(E3 / 128, Mrows / 128, 1), blk, 0, stream>>>(
      x_bf, 0, DIN, wqkv_bf, 0, DIN, (void*)qkv_bf, 0, E3, bqkv, 0, DIN);

  // 3) scores S = Q.K^T + bias  (per batch), fp32
  gemm_wmma<EPI_SCORES><<<dim3(Tn / 128, Tn / 128, Bn), blk, 0, stream>>>(
      qkv_bf, (long)Tn * E3, E3,
      qkv_bf + DEMB, (long)Tn * E3, E3,
      (void*)Sbuf, (long)Tn * Tn, Tn, bias, Tn, DEMB);

  // 4) row softmax -> bf16 P         5) transpose V -> V^T (K-major for P.V)
  softmax_rows<<<(unsigned)(Bn * Tn), 256, 0, stream>>>(Sbuf, Pbuf);
  transpose_v<<<dim3(Tn / 32, DEMB / 32, Bn), 256, 0, stream>>>(qkv_bf + 2 * DEMB, VTbuf);

  // 6) values = P.V^T^T  (both operands K-contiguous now), bf16
  gemm_wmma<EPI_PLAIN><<<dim3(DEMB / 128, Tn / 128, Bn), blk, 0, stream>>>(
      Pbuf, (long)Tn * Tn, Tn,
      VTbuf, (long)DEMB * Tn, Tn,
      (void*)Vbuf, (long)Tn * DEMB, DEMB, nullptr, 0, Tn);

  // 7) output projection -> fp32 d_out
  gemm_wmma<EPI_OUT><<<dim3(DOUT / 128, Mrows / 128, 1), blk, 0, stream>>>(
      Vbuf, 0, DEMB, wout_bf, 0, DEMB, d_out, 0, DOUT, bout, 0, DEMB);
}
